// DiffShareMamba_26388279066690
// MI455X (gfx1250) — matrix operand order repaired and checked by reading, not compiled, not run on hardware
//
#include <hip/hip_runtime.h>

// ---------------------------------------------------------------------------
// DiffShareMamba for MI455X (gfx1250, wave32, WMMA).
// Dense projections use a guard-free WMMA GEMM: weights are pre-packed once
// per launch into padded f16 buffers in exact WMMA B-fragment layout, and the
// dt-rank activation is zero-padded to K=32, so the GEMM inner loop is pure
// unconditional vector loads + v_wmma_f32_16x16x32_f16 (2 N-tiles per wave).
// The sequential selective scan uses 16 lanes per recurrence (one per SSM
// state) with shfl_xor reductions, unrolled x4 over b128 loads.
// ---------------------------------------------------------------------------

#define BSZ 4
#define CCH 96
#define HH 64
#define WW 64
#define LL (HH * WW)          // 4096
#define DN 192                // d_inner
#define NS 16                 // ssm state
#define RR 6                  // dt_rank
#define NX (RR + 2 * NS)      // 38
#define MLPD (4 * CCH)        // 384
#define EPSV 1e-5f

typedef __attribute__((ext_vector_type(16))) _Float16 v16h;
typedef __attribute__((ext_vector_type(8)))  float    v8f;

enum { ACT_NONE = 0, ACT_GELU = 1, ACT_SOFTPLUS = 2 };

__device__ __forceinline__ float siluf(float x) { return x / (1.f + __expf(-x)); }
__device__ __forceinline__ float geluf(float x) {
  // tanh approximation (jax.nn.gelu default)
  float x3 = x * x * x;
  return 0.5f * x * (1.f + tanhf(0.7978845608028654f * (x + 0.044715f * x3)));
}
__device__ __forceinline__ float softplusf(float x) {
  return fmaxf(x, 0.f) + log1pf(__expf(-fabsf(x)));
}

// ---------------------------------------------------------------------------
// Weight pre-pack: W(k,n) strided fp32 -> padded f16 in WMMA B-fragment order
//   out[w][nt][kt][lane][e],  kk = (e>>3)*16+(lane>>4)*8+(e&7), nn = lane&15
// Out-of-range (K,N) entries are zero. One-shot; edge handling lives here.
// ---------------------------------------------------------------------------
__global__ void k_pack_w(const float* __restrict__ W, long b_rk, long b_rn, long w_zb,
                         int nw, int K, int N, int ktiles, int ntp,
                         _Float16* __restrict__ out, int total)
{
  int i = blockIdx.x * blockDim.x + threadIdx.x;
  if (i >= total) return;
  int e = i & 15;
  int lane = (i >> 4) & 31;
  int rest = i >> 9;
  int kt = rest % ktiles; rest /= ktiles;
  int nt = rest % ntp;
  int w = rest / ntp;
  int kk = ((e >> 3) << 4) + ((lane >> 4) << 3) + (e & 7);
  int nn = lane & 15;
  int kg = kt * 32 + kk, ng = nt * 16 + nn;
  float v = 0.f;
  if (kg < K && ng < N)
    v = W[(long)w * w_zb + (long)kg * b_rk + (long)ng * b_rn];
  out[i] = (_Float16)v;
}

// ---------------------------------------------------------------------------
// Guard-free strided batched GEMM:
//   D[z] = act(A[z] @ Bpack[z%wmod] + bias[z%wmod]) + Cres[z]
// A: element(m,k) at A + z*a_zb + m*a_rm + k*a_rk (fp32 -> f16); K = ktiles*32
// Bpack: padded f16 fragments (see k_pack_w); bp_zb in halves per weight-batch.
// Block = 128 threads = 4 waves; wave owns one 16-row M tile x two 16-col
// N tiles (one A fragment feeds two v_wmma). M must be a multiple of 64.
// ---------------------------------------------------------------------------
__global__ __launch_bounds__(128)
void k_gemm_wmma(const float* __restrict__ A, long a_rm, long a_rk, long a_zb,
                 const _Float16* __restrict__ Bp, long bp_zb, int wmod, int ktiles,
                 const float* __restrict__ bias, long bias_zb,
                 const float* __restrict__ Cres, long c_rm, long c_rn, long c_zb,
                 float* __restrict__ D, long d_rm, long d_rn, long d_zb,
                 int M, int N, int act)
{
  int z  = blockIdx.z;
  int wz = z % wmod;
  A += (long)z * a_zb;
  if (bias) bias += (long)wz * bias_zb;
  if (Cres) Cres += (long)z * c_zb;
  D += (long)z * d_zb;

  const int lane = threadIdx.x & 31;
  const int wave = threadIdx.x >> 5;          // 0..3
  const int m0 = (blockIdx.x * 4 + wave) * 16;
  const int rlo = lane & 15;                  // row (A) / col (D)
  const int hgrp = lane >> 4;                 // lane-half group

  const v16h* __restrict__ bp = (const v16h*)(Bp + (long)wz * bp_zb);
  const long nb0 = (long)(2 * blockIdx.y) * ktiles * 32 + lane;
  const long nb1 = nb0 + (long)ktiles * 32;

  const bool a_vec = (a_rk == 1) && ((a_rm & 3) == 0);
  const long arow = (long)(m0 + rlo) * a_rm;

  v8f acc0 = {}, acc1 = {};

  for (int kt = 0; kt < ktiles; ++kt) {
    const int k0 = kt * 32;
    // --- A fragment (unconditional loads; 16-bit A 16x32 ISA layout) ---
    v16h af;
    if (a_vec) {
      const float* ap = A + arow + k0 + hgrp * 8;
      float4 q0 = *(const float4*)(ap);
      float4 q1 = *(const float4*)(ap + 4);
      float4 q2 = *(const float4*)(ap + 16);
      float4 q3 = *(const float4*)(ap + 20);
      af[0] = (_Float16)q0.x; af[1] = (_Float16)q0.y;
      af[2] = (_Float16)q0.z; af[3] = (_Float16)q0.w;
      af[4] = (_Float16)q1.x; af[5] = (_Float16)q1.y;
      af[6] = (_Float16)q1.z; af[7] = (_Float16)q1.w;
      af[8]  = (_Float16)q2.x; af[9]  = (_Float16)q2.y;
      af[10] = (_Float16)q2.z; af[11] = (_Float16)q2.w;
      af[12] = (_Float16)q3.x; af[13] = (_Float16)q3.y;
      af[14] = (_Float16)q3.z; af[15] = (_Float16)q3.w;
    } else {
      const float* ap = A + arow + (long)k0 * a_rk;
      const long s = a_rk;
#pragma unroll
      for (int e = 0; e < 8; ++e)
        af[e] = (_Float16)ap[(hgrp * 8 + e) * s];
#pragma unroll
      for (int e = 8; e < 16; ++e)
        af[e] = (_Float16)ap[(16 + hgrp * 8 + (e - 8)) * s];
    }
    // --- B fragments: one 32B vector read each, already f16 ---
    v16h bf0 = bp[nb0 + kt * 32];
    v16h bf1 = bp[nb1 + kt * 32];

    acc0 = __builtin_amdgcn_wmma_f32_16x16x32_f16(
        false, af, false, bf0, (short)0, acc0, false, false);
    acc1 = __builtin_amdgcn_wmma_f32_16x16x32_f16(
        false, af, false, bf1, (short)0, acc1, false, false);
  }

  // --- epilogue: VGPR r holds row m = 8*hgrp + r, col = (tile base)+rlo ---
#pragma unroll
  for (int t = 0; t < 2; ++t) {
    const int ncol = (2 * blockIdx.y + t) * 16 + rlo;
    if (ncol < N) {
      float bv = bias ? bias[ncol] : 0.f;
#pragma unroll
      for (int r = 0; r < 8; ++r) {
        long mrow = m0 + hgrp * 8 + r;
        float v = (t ? acc1[r] : acc0[r]) + bv;
        if (act == ACT_GELU)          v = geluf(v);
        else if (act == ACT_SOFTPLUS) v = softplusf(v);
        if (Cres) v += Cres[mrow * c_rm + (long)ncol * c_rn];
        D[mrow * d_rm + (long)ncol * d_rn] = v;
      }
    }
  }
}

// --------------------------- elementwise kernels ---------------------------

// NCHW -> NHWC (optionally a-b for the diff path)
__global__ void k_to_nhwc(const float* __restrict__ a, const float* __restrict__ bsub,
                          float* __restrict__ out)
{
  int i = blockIdx.x * blockDim.x + threadIdx.x;
  if (i >= BSZ * LL * CCH) return;
  int c = i % CCH, l = (i / CCH) % LL, b = i / (CCH * LL);
  long src = ((long)b * CCH + c) * LL + l;
  float v = a[src];
  if (bsub) v -= bsub[src];
  out[i] = v;
}

// LayerNorm over rows of `dim` (dim multiple of 32), wave per row.
// Optional gate: y *= silu(gate[row*gate_rs + c]).
__global__ void k_ln(const float* __restrict__ x, const float* __restrict__ g,
                     const float* __restrict__ bb, float* __restrict__ y,
                     int rows, int dim, const float* __restrict__ gate, int gate_rs)
{
  int wave = threadIdx.x >> 5, lane = threadIdx.x & 31;
  int row = blockIdx.x * (blockDim.x >> 5) + wave;
  if (row >= rows) return;
  const float* xr = x + (long)row * dim;
  int nch = dim >> 5;
  float vals[8], s = 0.f, q = 0.f;
  for (int j = 0; j < nch; ++j) {
    float v = xr[j * 32 + lane];
    vals[j] = v; s += v; q += v * v;
  }
  for (int o = 16; o > 0; o >>= 1) { s += __shfl_xor(s, o, 32); q += __shfl_xor(q, o, 32); }
  float mean = s / dim;
  float rstd = rsqrtf(q / dim - mean * mean + EPSV);
  float* yr = y + (long)row * dim;
  const float* gr = gate ? gate + (long)row * gate_rs : nullptr;
  for (int j = 0; j < nch; ++j) {
    int c = j * 32 + lane;
    float v = (vals[j] - mean) * rstd * g[c] + bb[c];
    if (gr) v *= siluf(gr[c]);
    yr[c] = v;
  }
}

// Depthwise 3x3 + bias + SiLU. Input = x-half of xz (NHWC rows of 2*DN),
// output NCHW (B,DN,H,W) ready for cross-scan.
__global__ void k_dwconv(const float* __restrict__ xz, const float* __restrict__ wgt,
                         const float* __restrict__ bias, float* __restrict__ out)
{
  int i = blockIdx.x * blockDim.x + threadIdx.x;
  if (i >= BSZ * DN * LL) return;
  int hw = i % LL, d = (i / LL) % DN, b = i / (LL * DN);
  int h = hw / WW, w = hw % WW;
  float acc = bias[d];
#pragma unroll
  for (int kh = 0; kh < 3; ++kh) {
    int hh = h + kh - 1;
    if (hh < 0 || hh >= HH) continue;
#pragma unroll
    for (int kw = 0; kw < 3; ++kw) {
      int ww2 = w + kw - 1;
      if (ww2 < 0 || ww2 >= WW) continue;
      acc += xz[((long)b * LL + hh * WW + ww2) * (2 * DN) + d] * wgt[d * 9 + kh * 3 + kw];
    }
  }
  out[i] = siluf(acc);
}

// Build the 4 traversal orders (B,4,DN,L) from (B,DN,H,W)
__global__ void k_cross_scan(const float* __restrict__ xc, float* __restrict__ xs)
{
  int i = blockIdx.x * blockDim.x + threadIdx.x;
  if (i >= BSZ * DN * LL) return;
  int l = i % LL, d = (i / LL) % DN, b = i / (LL * DN);
  float v = xc[i];
  int h = l / WW, w = l % WW;
  int lt = w * HH + h;
  long base = ((long)(b * 4) * DN + d) * LL;
  long kst = (long)DN * LL;
  xs[base + l] = v;
  xs[base + kst + lt] = v;
  xs[base + 2 * kst + (LL - 1 - l)] = v;
  xs[base + 3 * kst + (LL - 1 - lt)] = v;
}

// dts rows (R=6) of xdbl zero-padded to 32 rows: (B*4, 32, L)
__global__ void k_pad_dts(const float* __restrict__ xdbl, float* __restrict__ dts)
{
  int i = blockIdx.x * blockDim.x + threadIdx.x;
  if (i >= BSZ * 4 * 32 * LL) return;
  int l = i % LL, r = (i / LL) % 32, zz = i / (32 * LL);
  dts[i] = (r < RR) ? xdbl[((long)zz * NX + r) * LL + l] : 0.f;
}

// Selective scan: 16 lanes per (b,k,d) recurrence, one SSM state n per lane.
// y(l) reduced across the 16-lane group via shfl_xor; also fuses + Ds*x.
// Unrolled x4 over contiguous l so all streams use b128 loads.
__global__ __launch_bounds__(256)
void k_scan(const float* __restrict__ delta, const float* __restrict__ xs,
            const float* __restrict__ xdbl, const float* __restrict__ A_logs,
            const float* __restrict__ Ds, float* __restrict__ ys)
{
  int gid = blockIdx.x * (blockDim.x >> 4) + (threadIdx.x >> 4);
  int n = threadIdx.x & 15;
  if (gid >= BSZ * 4 * DN) return;            // exact grid; never taken
  int b = gid / (4 * DN), rem = gid % (4 * DN);
  int k = rem / DN, d = rem % DN;
  long chan = (long)(b * 4 + k) * DN + d;
  const float* dl = delta + chan * LL;
  const float* xl = xs + chan * LL;
  long xb = ((long)(b * 4 + k) * NX + RR) * LL;
  const float* Bl = xdbl + xb + (long)n * LL;
  const float* Cl = xdbl + xb + (long)(NS + n) * LL;
  float Ac = -__expf(A_logs[((long)k * DN + d) * NS + n]);
  float Dc = Ds[k * DN + d];
  float* yl = ys + chan * LL;
  float h = 0.f;
  for (int l = 0; l < LL; l += 4) {
    float4 dt4 = *(const float4*)(dl + l);
    float4 xt4 = *(const float4*)(xl + l);
    float4 B4  = *(const float4*)(Bl + l);
    float4 C4  = *(const float4*)(Cl + l);
    float4 y4;
    {
      h = h * __expf(dt4.x * Ac) + dt4.x * xt4.x * B4.x;
      float y = h * C4.x;
      y += __shfl_xor(y, 8, 32); y += __shfl_xor(y, 4, 32);
      y += __shfl_xor(y, 2, 32); y += __shfl_xor(y, 1, 32);
      y4.x = y + Dc * xt4.x;
    }
    {
      h = h * __expf(dt4.y * Ac) + dt4.y * xt4.y * B4.y;
      float y = h * C4.y;
      y += __shfl_xor(y, 8, 32); y += __shfl_xor(y, 4, 32);
      y += __shfl_xor(y, 2, 32); y += __shfl_xor(y, 1, 32);
      y4.y = y + Dc * xt4.y;
    }
    {
      h = h * __expf(dt4.z * Ac) + dt4.z * xt4.z * B4.z;
      float y = h * C4.z;
      y += __shfl_xor(y, 8, 32); y += __shfl_xor(y, 4, 32);
      y += __shfl_xor(y, 2, 32); y += __shfl_xor(y, 1, 32);
      y4.z = y + Dc * xt4.z;
    }
    {
      h = h * __expf(dt4.w * Ac) + dt4.w * xt4.w * B4.w;
      float y = h * C4.w;
      y += __shfl_xor(y, 8, 32); y += __shfl_xor(y, 4, 32);
      y += __shfl_xor(y, 2, 32); y += __shfl_xor(y, 1, 32);
      y4.w = y + Dc * xt4.w;
    }
    if (n == 0) *(float4*)(yl + l) = y4;
  }
}

// Invert the 4 scan orders; output channels-last (B,L,DN)
__global__ void k_cross_merge(const float* __restrict__ ys, float* __restrict__ out)
{
  int i = blockIdx.x * blockDim.x + threadIdx.x;
  if (i >= BSZ * LL * DN) return;
  int d = i % DN, l = (i / DN) % LL, b = i / (DN * LL);
  long cb = ((long)(b * 4) * DN + d) * LL;
  long kst = (long)DN * LL;
  int h = l / WW, w = l % WW;
  int lt = w * HH + h;
  out[i] = ys[cb + l] + ys[cb + 2 * kst + (LL - 1 - l)]
         + ys[cb + kst + lt] + ys[cb + 3 * kst + (LL - 1 - lt)];
}

// 3x3 conv (no bias) + eval BatchNorm (running stats 0/1) + optional residual + optional ReLU
__global__ void k_conv3x3(const float* __restrict__ in, const float* __restrict__ wgt,
                          const float* __restrict__ g, const float* __restrict__ bb,
                          int Cin, int Cout, const float* __restrict__ res, int relu,
                          float* __restrict__ out)
{
  int i = blockIdx.x * blockDim.x + threadIdx.x;
  if (i >= BSZ * Cout * LL) return;
  int hw = i % LL, co = (i / LL) % Cout, b = i / (LL * Cout);
  int h = hw / WW, w = hw % WW;
  float acc = 0.f;
  for (int ci = 0; ci < Cin; ++ci) {
    const float* ip = in + ((long)b * Cin + ci) * LL;
    const float* wp = wgt + ((long)co * Cin + ci) * 9;
#pragma unroll
    for (int kh = 0; kh < 3; ++kh) {
      int hh = h + kh - 1;
      if (hh < 0 || hh >= HH) continue;
#pragma unroll
      for (int kw = 0; kw < 3; ++kw) {
        int ww2 = w + kw - 1;
        if (ww2 < 0 || ww2 >= WW) continue;
        acc += ip[hh * WW + ww2] * wp[kh * 3 + kw];
      }
    }
  }
  float val = acc * (g[co] * rsqrtf(1.f + EPSV)) + bb[co];
  if (res) val += res[i];
  if (relu) val = fmaxf(val, 0.f);
  out[i] = val;
}

__global__ void k_add(const float* __restrict__ a, const float* __restrict__ b,
                      float* __restrict__ o, int n)
{
  int i = blockIdx.x * blockDim.x + threadIdx.x;
  if (i < n) o[i] = a[i] + b[i];
}

// ------------------------------- host side ---------------------------------

struct VSSP {
  const float *A_logs, *Ds, *conv_b, *conv_w, *dt_b, *dt_w, *fc1_b, *fc1_w,
              *fc2_b, *fc2_w, *in_w, *n1_b, *n1_g, *n2_b, *n2_g, *on_b, *on_g,
              *out_w, *xproj_w;
};

static VSSP load_vss(void* const* d_in, int base) {
  VSSP p;
  p.A_logs  = (const float*)d_in[base + 0];
  p.Ds      = (const float*)d_in[base + 1];
  p.conv_b  = (const float*)d_in[base + 2];
  p.conv_w  = (const float*)d_in[base + 3];
  p.dt_b    = (const float*)d_in[base + 4];
  p.dt_w    = (const float*)d_in[base + 5];
  p.fc1_b   = (const float*)d_in[base + 6];
  p.fc1_w   = (const float*)d_in[base + 7];
  p.fc2_b   = (const float*)d_in[base + 8];
  p.fc2_w   = (const float*)d_in[base + 9];
  p.in_w    = (const float*)d_in[base + 10];
  p.n1_b    = (const float*)d_in[base + 11];
  p.n1_g    = (const float*)d_in[base + 12];
  p.n2_b    = (const float*)d_in[base + 13];
  p.n2_g    = (const float*)d_in[base + 14];
  p.on_b    = (const float*)d_in[base + 15];
  p.on_g    = (const float*)d_in[base + 16];
  p.out_w   = (const float*)d_in[base + 17];
  p.xproj_w = (const float*)d_in[base + 18];
  return p;
}

// packed-weight tile counts: {ktiles, ntp(padded even)}
#define KT_IN   3
#define NT_IN   24
#define KT_XP   6
#define NT_XP   4      // N=38 -> 3 tiles, padded to 4
#define KT_DT   1      // K=6 zero-padded to 32
#define NT_DT   12
#define KT_OUT  6
#define NT_OUT  6
#define KT_FC1  3
#define NT_FC1  24
#define KT_FC2  12
#define NT_FC2  6

struct PKW { _Float16 *in, *xp, *dt, *out, *fc1, *fc2; };

struct WSB {
  float *t_x, *t_n, *t_xz, *t_xc, *t_xs, *t_xdbl, *t_dts, *t_delta, *t_ys, *t_y, *t_mlp;
};

static inline void packw(hipStream_t st, const float* W, long brk, long brn, long wzb,
                         int nw, int K, int N, int kt, int ntp, _Float16* out)
{
  int total = nw * ntp * kt * 512;
  k_pack_w<<<(total + 255) / 256, 256, 0, st>>>(W, brk, brn, wzb, nw, K, N, kt, ntp, out, total);
}

static inline void gemm(hipStream_t st,
                        const float* A, long arm, long ark, long azb,
                        const _Float16* Bp, long bpzb, int wmod, int ktiles, int ntp,
                        const float* bias, long biaszb,
                        const float* Cres, long crm, long crn, long czb,
                        float* D, long drm, long drn, long dzb,
                        int M, int N, int act, int batch)
{
  dim3 g(M / 64, ntp / 2, batch);
  k_gemm_wmma<<<g, 128, 0, st>>>(A, arm, ark, azb, Bp, bpzb, wmod, ktiles,
                                 bias, biaszb, Cres, crm, crn, czb,
                                 D, drm, drn, dzb, M, N, act);
}

static void run_vss(hipStream_t st, const WSB& w, const PKW& pk,
                    const float* in1, const float* in2,
                    const VSSP& p, float* outD, long out_zb)
{
  const int T = 256;
  const int nBLC = BSZ * LL * CCH;

  // permute to channels-last (and optional subtract)
  k_to_nhwc<<<(nBLC + T - 1) / T, T, 0, st>>>(in1, in2, w.t_x);
  // LN1
  k_ln<<<(BSZ * LL) / 4, 128, 0, st>>>(w.t_x, p.n1_g, p.n1_b, w.t_n, BSZ * LL, CCH, nullptr, 0);
  // in_proj: (B*L,96) @ (96,384) -> xz
  gemm(st, w.t_n, CCH, 1, 0, pk.in, 0, 1, KT_IN, NT_IN, nullptr, 0,
       nullptr, 0, 0, 0, w.t_xz, 2 * DN, 1, 0, BSZ * LL, 2 * DN, ACT_NONE, 1);
  // depthwise conv + silu
  k_dwconv<<<(BSZ * DN * LL + T - 1) / T, T, 0, st>>>(w.t_xz, p.conv_w, p.conv_b, w.t_xc);
  // cross-scan
  k_cross_scan<<<(BSZ * DN * LL + T - 1) / T, T, 0, st>>>(w.t_xc, w.t_xs);
  // x_proj: per (b,k): (L,192) @ (192,38); weights repeat every 4 batches
  gemm(st, w.t_xs, 1, LL, (long)DN * LL,
       pk.xp, (long)NT_XP * KT_XP * 512, 4, KT_XP, NT_XP, nullptr, 0,
       nullptr, 0, 0, 0, w.t_xdbl, 1, LL, (long)NX * LL,
       LL, NX, ACT_NONE, BSZ * 4);
  // zero-pad dts rows to K=32
  k_pad_dts<<<(BSZ * 4 * 32 * LL + T - 1) / T, T, 0, st>>>(w.t_xdbl, w.t_dts);
  // dt_proj + bias + softplus: (L,32pad) @ (32pad,192)
  gemm(st, w.t_dts, 1, LL, (long)32 * LL,
       pk.dt, (long)NT_DT * KT_DT * 512, 4, KT_DT, NT_DT, p.dt_b, DN,
       nullptr, 0, 0, 0, w.t_delta, 1, LL, (long)DN * LL,
       LL, DN, ACT_SOFTPLUS, BSZ * 4);
  // selective scan (fuses + Ds*xs)
  k_scan<<<(BSZ * 4 * DN * 16) / 256, 256, 0, st>>>(w.t_delta, w.t_xs, w.t_xdbl,
                                                    p.A_logs, p.Ds, w.t_ys);
  // cross-merge -> (B,L,DN)
  k_cross_merge<<<(BSZ * LL * DN + T - 1) / T, T, 0, st>>>(w.t_ys, w.t_y);
  // out-norm + silu(z) gate (in place; z is second half of xz rows)
  k_ln<<<(BSZ * LL) / 4, 128, 0, st>>>(w.t_y, p.on_g, p.on_b, w.t_y, BSZ * LL, DN,
                                       w.t_xz + DN, 2 * DN);
  // out_proj + residual: t_x = t_x + y @ out_w
  gemm(st, w.t_y, DN, 1, 0, pk.out, 0, 1, KT_OUT, NT_OUT, nullptr, 0,
       w.t_x, CCH, 1, 0, w.t_x, CCH, 1, 0, BSZ * LL, CCH, ACT_NONE, 1);
  // LN2 -> t_n
  k_ln<<<(BSZ * LL) / 4, 128, 0, st>>>(w.t_x, p.n2_g, p.n2_b, w.t_n, BSZ * LL, CCH, nullptr, 0);
  // fc1 + gelu
  gemm(st, w.t_n, CCH, 1, 0, pk.fc1, 0, 1, KT_FC1, NT_FC1, p.fc1_b, 0,
       nullptr, 0, 0, 0, w.t_mlp, MLPD, 1, 0, BSZ * LL, MLPD, ACT_GELU, 1);
  // fc2 + residual, write NCHW (batched over b so D can use custom batch stride)
  gemm(st, w.t_mlp, MLPD, 1, (long)LL * MLPD, pk.fc2, 0, 1, KT_FC2, NT_FC2, p.fc2_b, 0,
       w.t_x, CCH, 1, (long)LL * CCH, outD, 1, LL, out_zb,
       LL, CCH, ACT_NONE, BSZ);
}

static void pack_all(hipStream_t st, const VSSP& p, const PKW& pk)
{
  packw(st, p.in_w,    2 * DN, 1, 0,              1, CCH,  2 * DN, KT_IN,  NT_IN,  pk.in);
  packw(st, p.xproj_w, 1, DN,  (long)NX * DN,     4, DN,   NX,     KT_XP,  NT_XP,  pk.xp);
  packw(st, p.dt_w,    1, RR,  (long)DN * RR,     4, RR,   DN,     KT_DT,  NT_DT,  pk.dt);
  packw(st, p.out_w,   CCH, 1, 0,                 1, DN,   CCH,    KT_OUT, NT_OUT, pk.out);
  packw(st, p.fc1_w,   MLPD, 1, 0,                1, CCH,  MLPD,   KT_FC1, NT_FC1, pk.fc1);
  packw(st, p.fc2_w,   CCH, 1, 0,                 1, MLPD, CCH,    KT_FC2, NT_FC2, pk.fc2);
}

extern "C" void kernel_launch(void* const* d_in, const int* in_sizes, int n_in,
                              void* d_out, int out_size, void* d_ws, size_t ws_size,
                              hipStream_t stream)
{
  (void)in_sizes; (void)n_in; (void)out_size; (void)ws_size;

  const float* m1 = (const float*)d_in[0];
  const float* m2 = (const float*)d_in[1];
  VSSP diffp  = load_vss(d_in, 2);
  const float* bn1_b = (const float*)d_in[21];
  const float* bn1_g = (const float*)d_in[22];
  const float* bn2_b = (const float*)d_in[23];
  const float* bn2_g = (const float*)d_in[24];
  const float* c1_w  = (const float*)d_in[25];
  const float* c2_w  = (const float*)d_in[26];
  VSSP sharep = load_vss(d_in, 27);
  const float* share_bn_b   = (const float*)d_in[46];
  const float* share_bn_g   = (const float*)d_in[47];
  const float* share_conv_w = (const float*)d_in[48];

  // ---- workspace carving (256B aligned) ----
  size_t off = 0;
  auto allocb = [&](size_t nbytes) {
    void* p = (char*)d_ws + off;
    off += (nbytes + 255) & ~(size_t)255;
    return p;
  };
  auto allocf = [&](size_t nf) { return (float*)allocb(nf * sizeof(float)); };
  auto alloch = [&](size_t nh) { return (_Float16*)allocb(nh * sizeof(_Float16)); };

  const size_t nBLC = (size_t)BSZ * LL * CCH;
  float* sm_concat = allocf((size_t)BSZ * 2 * CCH * LL);  // [share_m1 | share_m2] per batch
  float* diff_out  = allocf(nBLC);
  float* share_m   = allocf(nBLC);
  float* fuse_in   = allocf(nBLC);
  float* r1        = allocf(nBLC);
  WSB w;
  w.t_x    = allocf(nBLC);
  w.t_n    = allocf(nBLC);
  w.t_xz   = allocf((size_t)BSZ * LL * 2 * DN);
  w.t_xc   = allocf((size_t)BSZ * DN * LL);
  w.t_xs   = allocf((size_t)BSZ * 4 * DN * LL);
  w.t_xdbl = allocf((size_t)BSZ * 4 * NX * LL);
  w.t_dts  = allocf((size_t)BSZ * 4 * 32 * LL);
  w.t_delta= allocf((size_t)BSZ * 4 * DN * LL);
  w.t_ys   = allocf((size_t)BSZ * 4 * DN * LL);
  w.t_y    = allocf((size_t)BSZ * LL * DN);
  w.t_mlp  = allocf((size_t)BSZ * LL * MLPD);

  PKW pk_sh, pk_df;
  pk_sh.in  = alloch((size_t)NT_IN  * KT_IN  * 512);
  pk_sh.xp  = alloch((size_t)4 * NT_XP * KT_XP * 512);
  pk_sh.dt  = alloch((size_t)4 * NT_DT * KT_DT * 512);
  pk_sh.out = alloch((size_t)NT_OUT * KT_OUT * 512);
  pk_sh.fc1 = alloch((size_t)NT_FC1 * KT_FC1 * 512);
  pk_sh.fc2 = alloch((size_t)NT_FC2 * KT_FC2 * 512);
  pk_df.in  = alloch((size_t)NT_IN  * KT_IN  * 512);
  pk_df.xp  = alloch((size_t)4 * NT_XP * KT_XP * 512);
  pk_df.dt  = alloch((size_t)4 * NT_DT * KT_DT * 512);
  pk_df.out = alloch((size_t)NT_OUT * KT_OUT * 512);
  pk_df.fc1 = alloch((size_t)NT_FC1 * KT_FC1 * 512);
  pk_df.fc2 = alloch((size_t)NT_FC2 * KT_FC2 * 512);

  // ---- pre-pack all weights into WMMA fragment layout ----
  pack_all(stream, sharep, pk_sh);
  pack_all(stream, diffp, pk_df);

  // ---- three VSS blocks (scratch reused sequentially) ----
  run_vss(stream, w, pk_sh, m1, nullptr, sharep, sm_concat,            (long)2 * CCH * LL);
  run_vss(stream, w, pk_sh, m2, nullptr, sharep, sm_concat + (long)CCH * LL, (long)2 * CCH * LL);
  run_vss(stream, w, pk_df, m1, m2,      diffp,  diff_out,             (long)CCH * LL);

  // ---- fusion: share conv (192->96) + BN + ReLU ----
  const int T = 256;
  const int nOut = BSZ * CCH * LL;
  k_conv3x3<<<(nOut + T - 1) / T, T, 0, stream>>>(sm_concat, share_conv_w,
      share_bn_g, share_bn_b, 2 * CCH, CCH, nullptr, 1, share_m);
  // share_m + diff_m
  k_add<<<(nOut + T - 1) / T, T, 0, stream>>>(share_m, diff_out, fuse_in, nOut);
  // residual block
  k_conv3x3<<<(nOut + T - 1) / T, T, 0, stream>>>(fuse_in, c1_w, bn1_g, bn1_b,
      CCH, CCH, nullptr, 1, r1);
  k_conv3x3<<<(nOut + T - 1) / T, T, 0, stream>>>(r1, c2_w, bn2_g, bn2_b,
      CCH, CCH, fuse_in, 1, (float*)d_out);
}
